// Region_26714696581523
// MI455X (gfx1250) — compile-verified, matrix operand order checked by reading, not run on hardware
//
#include <hip/hip_runtime.h>
#include <stdint.h>

// ---------------------------------------------------------------- types
typedef unsigned short bh; // bf16 bit pattern
typedef __attribute__((ext_vector_type(16))) unsigned short v16u;
typedef __attribute__((ext_vector_type(16))) __bf16         v16bf;
typedef __attribute__((ext_vector_type(8)))  float          v8f;

#define DEV __device__ __forceinline__

DEV bh f2bh(float f) {
  union { float f; uint32_t u; } c; c.f = f;
  uint32_t u = c.u;
  uint32_t r = u + 0x7fffu + ((u >> 16) & 1u); // RNE
  return (bh)(r >> 16);
}
DEV float eluf(float x)  { return x > 0.f ? x : __expf(x) - 1.f; }
DEV float seluf(float x) {
  const float A = 1.6732632423543772f, S = 1.0507009873554805f;
  return x > 0.f ? S * x : S * A * (__expf(x) - 1.f);
}
DEV float sigf(float x)  { return 1.f / (1.f + __expf(-x)); }
DEV float reluf(float x) { return x > 0.f ? x : 0.f; }

__constant__ int d_glist[7] = {5, 9, 9, 25, 9, 9, 12};
__constant__ int d_ghead[7] = {0, 5, 14, 23, 48, 57, 66};

// ---------------------------------------------------------------- misc
__global__ void k_zero1(float* p) { if (threadIdx.x == 0 && blockIdx.x == 0) p[0] = 0.f; }

// W (K x N row-major) -> out (N x Kpad row-major, bf16, zero-padded K tail)
__global__ void k_cvt_w_t(const float* __restrict__ W, bh* __restrict__ out,
                          int K, int N, int Kpad) {
  long n = (long)N * Kpad;
  long i  = (long)blockIdx.x * blockDim.x + threadIdx.x;
  long st = (long)gridDim.x * blockDim.x;
  for (; i < n; i += st) {
    int nn = (int)(i / Kpad), k = (int)(i % Kpad);
    out[i] = (k < K) ? f2bh(W[(long)k * N + nn]) : (bh)0;
  }
}
// W (N x K row-major, e.g. emb) -> out (N x Kpad row-major, bf16, zero-padded)
__global__ void k_cvt_w_pad(const float* __restrict__ W, bh* __restrict__ out,
                            int K, int N, int Kpad) {
  long n = (long)N * Kpad;
  long i  = (long)blockIdx.x * blockDim.x + threadIdx.x;
  long st = (long)gridDim.x * blockDim.x;
  for (; i < n; i += st) {
    int nn = (int)(i / Kpad), k = (int)(i % Kpad);
    out[i] = (k < K) ? f2bh(W[(long)nn * K + k]) : (bh)0;
  }
}

// ---------------------------------------------------------------- WMMA GEMM
// C[M,N] = A[M,Kpad](bf16, zero-padded) @ BT[N,Kpad](bf16, zero-padded) (+bias)
// Requirements (guaranteed by the pipeline): M % 64 == 0, Kpad % 32 == 0,
// A rows and BT rows 64B-aligned. No K/M bounds checks -> staging is pure
// global_load_b128 + ds_store_b128; WMMA loop branch-free, EXEC all-ones.
__global__ __launch_bounds__(128)
void k_gemm_wmma(const bh* __restrict__ A, int lda,
                 const bh* __restrict__ BT, int ldb,
                 const float* __restrict__ bias,
                 float* __restrict__ C, int ldc,
                 bh* __restrict__ Cbf, int ldcbf,
                 int N, int Kpad) {
  constexpr int LDT = 40; // padded LDS stride in halves (80B: 16B-aligned, bank-spread)
  __shared__ __align__(16) bh As[64 * LDT];
  __shared__ __align__(16) bh BsT[16 * LDT];

  const int tid  = threadIdx.x;
  const int lane = tid & 31;
  const int wave = tid >> 5;
  const int mBase = blockIdx.x * 64;
  const int n0    = blockIdx.y * 16;

  v8f acc;
#pragma unroll
  for (int i = 0; i < 8; i++) acc[i] = 0.f;

  const int rowL = wave * 16 + (lane & 15); // fragment A row for this lane
  const int nL   = lane & 15;               // fragment B col for this lane
  const int kg   = (lane >> 4) * 8;         // K-group (ISA 16-bit operand layout)

  // staging assignments: A = 64 rows x 32 halves (2x b128 per thread)
  const int ar = tid >> 1;
  const int ah = (tid & 1) * 16;
  // B = 16 rows x 32 halves (1x b128 per thread, tid < 64)
  const int br = tid >> 2;
  const int bq = (tid & 3) * 8;
  const bool bAct = tid < 64;
  const bool bIn  = bAct && (n0 + br < N);

  const bh* Arow = A + (long)(mBase + ar) * lda + ah;
  const bh* Brow = BT + (long)(n0 + br) * ldb + bq;

  for (int k0 = 0; k0 < Kpad; k0 += 32) {
    if (k0 + 32 < Kpad)
      __builtin_prefetch((const void*)(Arow + k0 + 32), 0, 1); // global_prefetch_b8
    // issue global loads before the barrier: overlap with previous WMMA
    uint4 a0 = *(const uint4*)(Arow + k0);
    uint4 a1 = *(const uint4*)(Arow + k0 + 8);
    uint4 b0 = make_uint4(0u, 0u, 0u, 0u);
    if (bIn) b0 = *(const uint4*)(Brow + k0);

    __syncthreads(); // previous fragment reads done
    *(uint4*)&As[ar * LDT + ah]     = a0;
    *(uint4*)&As[ar * LDT + ah + 8] = a1;
    if (bAct) *(uint4*)&BsT[br * LDT + bq] = b0;
    __syncthreads();

    union { v16u v; uint4 q[2]; } fa, fb;
    fa.q[0] = *(const uint4*)&As[rowL * LDT + kg];       // k = kg..kg+7
    fa.q[1] = *(const uint4*)&As[rowL * LDT + 16 + kg];  // k = 16+kg..
    fb.q[0] = *(const uint4*)&BsT[nL * LDT + kg];
    fb.q[1] = *(const uint4*)&BsT[nL * LDT + 16 + kg];

    acc = __builtin_amdgcn_wmma_f32_16x16x32_bf16(
        false, __builtin_bit_cast(v16bf, fa.v),
        false, __builtin_bit_cast(v16bf, fb.v),
        (short)0, acc, false, false);
  }

  const int m0 = mBase + wave * 16;
  const int mr = m0 + ((lane >> 4) ? 8 : 0);
  const int n  = n0 + nL;
  float badd = (bias != nullptr && n < N) ? bias[n] : 0.f;
#pragma unroll
  for (int r = 0; r < 8; r++) {
    int m = mr + r;
    float v = acc[r] + badd;
    if (C != nullptr && n < N) C[(long)m * ldc + n] = v;
    if (Cbf != nullptr) Cbf[(long)m * ldcbf + n] = (n < N) ? f2bh(v) : (bh)0;
  }
}

// ---------------------------------------------------------------- get_adj front
// o = P@x + bias ; t = o@Q ; g = elu(t@theta)    (block per sample)
__global__ void k_ga_front(const float* __restrict__ xin, int sampStride, int off, int c, int f,
                           const float* __restrict__ P, const float* __restrict__ bias,
                           const float* __restrict__ Q, const float* __restrict__ theta,
                           float* __restrict__ g) {
  __shared__ float so[352], st[352];
  const int tid = threadIdx.x, nt = blockDim.x;
  const long b = blockIdx.x;
  const int cf = c * f;
  const float* xb = xin + b * (long)sampStride + (long)off * f;
  for (int e = tid; e < cf; e += nt) {
    int i = e / f, m = e % f;
    float s = bias[e];
    for (int n2 = 0; n2 < c; n2++) s += P[i * c + n2] * xb[n2 * f + m];
    so[e] = s;
  }
  __syncthreads();
  for (int e = tid; e < cf; e += nt) {
    int i = e / f, k = e % f;
    float s = 0.f;
    for (int m = 0; m < f; m++) s += so[i * f + m] * Q[m * f + k];
    st[e] = s;
  }
  __syncthreads();
  const long tot = (long)c * cf;
  float* gb = g + b * tot;
  for (long e = tid; e < tot; e += nt) {
    int i = (int)(e / cf), col = (int)(e % cf);
    float s = 0.f;
    for (int k = 0; k < f; k++) s += st[i * f + k] * theta[(long)k * cf + col];
    gb[e] = eluf(s);
  }
}

// adj_bf16[b,fp] row (ccPad halves, zero-padded) = bf16(normalize(g[b,:, :,fp]))
__global__ void k_transpose_norm(const float* __restrict__ g, bh* __restrict__ adjbf,
                                 int c, int f, int ccPad) {
  __shared__ float sAx[625];
  __shared__ float sdi[32];
  const int tid = threadIdx.x, nt = blockDim.x;
  const long bf = blockIdx.x;
  const long b = bf / f;
  const int fp = (int)(bf % f);
  const int cc = c * c, cf = c * f;
  const float* gb = g + b * (long)c * cf;
  for (int e = tid; e < cc; e += nt) {
    int i = e / c, j = e % c;
    sAx[e] = gb[(long)i * cf + j * f + fp];
  }
  __syncthreads();
  for (int i = tid; i < c; i += nt) {
    float d = 1e-6f;
    for (int j = 0; j < c; j++) d += fabsf(sAx[i * c + j]);
    sdi[i] = rsqrtf(d);
  }
  __syncthreads();
  bh* ab = adjbf + bf * (long)ccPad;
  for (int e = tid; e < ccPad; e += nt) {
    if (e < cc) {
      int i = e / c, j = e % c;
      ab[e] = f2bh(sAx[e] * sdi[i] * sdi[j]);
    } else {
      ab[e] = (bh)0;
    }
  }
}

// ---------------------------------------------------------------- VQ pieces
__global__ void k_e2(const float* __restrict__ emb, float* __restrict__ e2, int Kcb, int D) {
  int n = blockIdx.x * blockDim.x + threadIdx.x;
  if (n < Kcb) {
    float s = 0.f;
    for (int d = 0; d < D; d++) { float v = emb[(long)n * D + d]; s += v * v; }
    e2[n] = s;
  }
}

__global__ void k_argmin(const float* __restrict__ S, const float* __restrict__ e2,
                         int* __restrict__ idx, long M, int Kcb) {
  long m = (long)blockIdx.x * blockDim.x + threadIdx.x;
  if (m >= M) return;
  const float* sm = S + m * Kcb;
  float best = e2[0] - 2.f * sm[0];
  int bi = 0;
  for (int n = 1; n < Kcb; n++) {
    float v = e2[n] - 2.f * sm[n];
    if (v < best) { best = v; bi = n; }
  }
  idx[m] = bi;
}

// gather zq, accumulate (1+beta)*mean((ze-zq)^2), emit zq bf16 (Dpad stride, 0-pad)
__global__ void k_gather_loss(const float* __restrict__ ze, const float* __restrict__ emb,
                              const int* __restrict__ idx, bh* __restrict__ zqbf,
                              float* __restrict__ lossAcc, float lscale,
                              long M, int D, int Dpad) {
  __shared__ float red[256];
  const long N = M * (long)Dpad;
  long i  = (long)blockIdx.x * blockDim.x + threadIdx.x;
  long st = (long)gridDim.x * blockDim.x;
  float local = 0.f;
  for (long e = i; e < N; e += st) {
    long m = e / Dpad; int d = (int)(e % Dpad);
    if (d < D) {
      float zq = emb[(long)idx[m] * D + d];
      float z  = ze[m * D + d];
      float df = z - zq;
      local += df * df;
      zqbf[e] = f2bh(zq);
    } else {
      zqbf[e] = (bh)0;
    }
  }
  red[threadIdx.x] = local;
  __syncthreads();
  for (int s = blockDim.x / 2; s > 0; s >>= 1) {
    if ((int)threadIdx.x < s) red[threadIdx.x] += red[threadIdx.x + s];
    __syncthreads();
  }
  if (threadIdx.x == 0) atomicAdd(lossAcc, red[0] * lscale);
}

// SE block + residual + elu, in place on recon (B,f,c,c)   (block per sample)
__global__ void k_se_resid(float* __restrict__ recon,
                           const float* __restrict__ w1, const float* __restrict__ b1,
                           const float* __restrict__ w2, const float* __restrict__ b2,
                           int f, int c, int h) {
  __shared__ float ss[64], sh[32], sg[64];
  const int tid = threadIdx.x, nt = blockDim.x;
  const int cc = c * c;
  const long n = (long)f * cc;
  float* rb = recon + (long)blockIdx.x * n;
  for (int i = tid; i < f; i += nt) ss[i] = 0.f;
  __syncthreads();
  for (long e = tid; e < n; e += nt) atomicAdd(&ss[(int)(e / cc)], rb[e]);
  __syncthreads();
  for (int j = tid; j < h; j += nt) {
    float s = b1[j];
    for (int i2 = 0; i2 < f; i2++) s += (ss[i2] / (float)cc) * w1[i2 * h + j];
    sh[j] = reluf(s);
  }
  __syncthreads();
  for (int i2 = tid; i2 < f; i2 += nt) {
    float s = b2[i2];
    for (int j = 0; j < h; j++) s += sh[j] * w2[j * f + i2];
    sg[i2] = sigf(s);
  }
  __syncthreads();
  for (long e = tid; e < n; e += nt) {
    float r = rb[e];
    rb[e] = eluf(r * sg[(int)(e / cc)] + r);
  }
}

// adj_out = normalize(elu(sum_f recon))   (block per sample)
__global__ void k_adjout(const float* __restrict__ recon, float* __restrict__ adjout,
                         int f, int c) {
  __shared__ float sAcc[625];
  __shared__ float sdi[32];
  const int tid = threadIdx.x, nt = blockDim.x;
  const int cc = c * c;
  const float* rb = recon + (long)blockIdx.x * f * cc;
  for (int e = tid; e < cc; e += nt) {
    float s = 0.f;
    for (int fp = 0; fp < f; fp++) s += rb[(long)fp * cc + e];
    sAcc[e] = eluf(s);
  }
  __syncthreads();
  for (int i = tid; i < c; i += nt) {
    float d = 1e-6f;
    for (int j = 0; j < c; j++) d += fabsf(sAcc[i * c + j]);
    sdi[i] = rsqrtf(d);
  }
  __syncthreads();
  float* ob = adjout + (long)blockIdx.x * cc;
  for (int e = tid; e < cc; e += nt) {
    int i = e / c, j = e % c;
    ob[e] = sAcc[e] * sdi[i] * sdi[j];
  }
}

// ---------------------------------------------------------------- CBAM in LDS
DEV void cbam_lds(float* X, int ldx, int c, int d,
                  const float* mw1, const float* mw2, const float* conv, int h,
                  float* scr) {
  const int tid = threadIdx.x, nt = blockDim.x;
  float *mv = scr, *xv = scr + 32, *hm = scr + 64, *hx = scr + 96,
        *ca = scr + 128, *spm = scr + 160, *spx = scr + 224, *sa = scr + 288;
  for (int i = tid; i < c; i += nt) {
    float s = 0.f, mx = -3.4e38f;
    for (int dd = 0; dd < d; dd++) { float v = X[i * ldx + dd]; s += v; mx = fmaxf(mx, v); }
    mv[i] = s / (float)d; xv[i] = mx;
  }
  __syncthreads();
  for (int j = tid; j < h; j += nt) {
    float s1 = 0.f, s2 = 0.f;
    for (int i = 0; i < c; i++) { s1 += mv[i] * mw1[i * h + j]; s2 += xv[i] * mw1[i * h + j]; }
    hm[j] = reluf(s1); hx[j] = reluf(s2);
  }
  __syncthreads();
  for (int i = tid; i < c; i += nt) {
    float s = 0.f;
    for (int j = 0; j < h; j++) s += (hm[j] + hx[j]) * mw2[j * c + i];
    ca[i] = sigf(s);
  }
  __syncthreads();
  for (int e = tid; e < c * d; e += nt) { int i = e / d, dd = e % d; X[i * ldx + dd] *= ca[i]; }
  __syncthreads();
  for (int dd = tid; dd < d; dd += nt) {
    float s = 0.f, mx = -3.4e38f;
    for (int i = 0; i < c; i++) { float v = X[i * ldx + dd]; s += v; mx = fmaxf(mx, v); }
    spm[dd] = s / (float)c; spx[dd] = mx;
  }
  __syncthreads();
  for (int dd = tid; dd < d; dd += nt) {
    float s = 0.f;
    for (int t = 0; t < 7; t++) {
      int p = dd - 3 + t;
      if (p >= 0 && p < d) s += conv[t] * spm[p] + conv[7 + t] * spx[p];
    }
    sa[dd] = sigf(s);
  }
  __syncthreads();
  for (int e = tid; e < c * d; e += nt) { int i = e / d, dd = e % d; X[i * ldx + dd] *= sa[dd]; }
  __syncthreads();
}

// ---------------------------------------------------------------- res_gcn (regions)
__global__ void k_res_gcn(const float* __restrict__ xin, int off, int c,
                          const float* __restrict__ adj,
                          const float* __restrict__ w1, const float* __restrict__ w2,
                          const float* __restrict__ m1w1, const float* __restrict__ m1w2,
                          const float* __restrict__ m1c,
                          const float* __restrict__ m2w1, const float* __restrict__ m2w2,
                          const float* __restrict__ m2c,
                          int h, float* __restrict__ feats) {
  __shared__ float sA[625], sX[1280], sT[800], scr[352];
  const int tid = threadIdx.x, nt = blockDim.x;
  const long b = blockIdx.x;
  const int cc = c * c;
  for (int e = tid; e < cc; e += nt) sA[e] = adj[b * (long)cc + e];
  for (int e = tid; e < c * 5; e += nt) {
    int i = e / 5, m = e % 5;
    sX[i * 50 + m] = xin[b * 390 + (long)(off + i) * 5 + m];
  }
  __syncthreads();
  for (int e = tid; e < c * 15; e += nt) {
    int i = e / 15, o = e % 15; float s = 0.f;
    for (int k = 0; k < 5; k++) s += sX[i * 50 + k] * w1[k * 15 + o];
    sT[e] = s;
  }
  __syncthreads();
  for (int e = tid; e < c * 15; e += nt) {
    int i = e / 15, o = e % 15; float s = 0.f;
    for (int j = 0; j < c; j++) s += sA[i * c + j] * sT[j * 15 + o];
    sX[i * 50 + 5 + o] = seluf(s);
  }
  __syncthreads();
  cbam_lds(sX, 50, c, 20, m1w1, m1w2, m1c, h, scr);
  for (int e = tid; e < c * 30; e += nt) {
    int i = e / 30, o = e % 30; float s = 0.f;
    for (int k = 0; k < 20; k++) s += sX[i * 50 + k] * w2[k * 30 + o];
    sT[e] = s;
  }
  __syncthreads();
  for (int e = tid; e < c * 30; e += nt) {
    int i = e / 30, o = e % 30; float s = 0.f;
    for (int j = 0; j < c; j++) s += sA[i * c + j] * sT[j * 30 + o];
    sX[i * 50 + 20 + o] = seluf(s);
  }
  __syncthreads();
  cbam_lds(sX, 50, c, 50, m2w1, m2w2, m2c, h, scr);
  for (int e = tid; e < c * 50; e += nt) {
    int i = e / 50, dd = e % 50;
    feats[(b * 78 + off + i) * 50 + dd] = eluf(sX[i * 50 + dd]);
  }
}

// ---------------------------------------------------------------- attention coarsen
__global__ void k_att_coarsen(const float* __restrict__ feats, const float* __restrict__ w,
                              float* __restrict__ coarse) {
  __shared__ float sfw[25 * 80];
  __shared__ float scol[80];
  __shared__ float satt[32];
  __shared__ float ssm[2];
  const int tid = threadIdx.x, nt = blockDim.x;
  const long b = blockIdx.x;
  for (int gph = 0; gph < 7; gph++) {
    int n = d_glist[gph], head = d_ghead[gph];
    for (int e = tid; e < n * 80; e += nt) {
      int i = e / 80, hh = e % 80; float s = 0.f;
      for (int k = 0; k < 50; k++) s += feats[(b * 78 + head + i) * 50 + k] * w[k * 80 + hh];
      sfw[e] = s;
    }
    __syncthreads();
    for (int hh = tid; hh < 80; hh += nt) {
      float s = 0.f;
      for (int i = 0; i < n; i++) s += sfw[i * 80 + hh];
      scol[hh] = s;
    }
    __syncthreads();
    for (int j = tid; j < n; j += nt) {
      float s = 0.f;
      for (int hh = 0; hh < 80; hh++) s += sfw[j * 80 + hh] * scol[hh];
      satt[j] = s;
    }
    __syncthreads();
    if (tid == 0) {
      float mx = satt[0];
      for (int j = 1; j < n; j++) mx = fmaxf(mx, satt[j]);
      float sm = 0.f;
      for (int j = 0; j < n; j++) { satt[j] = __expf(satt[j] - mx); sm += satt[j]; }
      ssm[0] = sm;
    }
    __syncthreads();
    for (int dd = tid; dd < 50; dd += nt) {
      float s = 0.f;
      for (int j = 0; j < n; j++) s += satt[j] * feats[(b * 78 + head + j) * 50 + dd];
      coarse[(b * 7 + gph) * 50 + dd] = s / ssm[0];
    }
    __syncthreads();
  }
}

// ---------------------------------------------------------------- res_gcn2 (coarse)
__global__ void k_res_gcn2(const float* __restrict__ coarse, const float* __restrict__ adj,
                           const float* __restrict__ w1, const float* __restrict__ w2,
                           const float* __restrict__ m1w1, const float* __restrict__ m1w2,
                           const float* __restrict__ m1c,
                           const float* __restrict__ m2w1, const float* __restrict__ m2w2,
                           const float* __restrict__ m2c,
                           float* __restrict__ out) {
  __shared__ float sA[49], sX0[7 * 50], sXa[7 * 64], sXo[7 * 64], sT[7 * 64], scr[352];
  const int tid = threadIdx.x, nt = blockDim.x;
  const long b = blockIdx.x;
  for (int e = tid; e < 49; e += nt) sA[e] = adj[b * 49 + e];
  for (int e = tid; e < 350; e += nt) sX0[e] = coarse[b * 350 + e];
  __syncthreads();
  for (int e = tid; e < 7 * 55; e += nt) {
    int i = e / 55, o = e % 55; float s = 0.f;
    for (int k = 0; k < 50; k++) s += sX0[i * 50 + k] * w1[k * 55 + o];
    sT[i * 64 + o] = s;
  }
  __syncthreads();
  for (int e = tid; e < 7 * 55; e += nt) {
    int i = e / 55, o = e % 55; float s = 0.f;
    for (int j = 0; j < 7; j++) s += sA[i * 7 + j] * sT[j * 64 + o];
    float v = seluf(s);
    sXa[i * 64 + o] = v; sXo[i * 64 + o] = v;
  }
  __syncthreads();
  cbam_lds(sXa, 64, 7, 55, m1w1, m1w2, m1c, 3, scr);
  for (int e = tid; e < 7 * 55; e += nt) { int i = e / 55, o = e % 55; sXa[i * 64 + o] += sXo[i * 64 + o]; }
  __syncthreads();
  for (int e = tid; e < 7 * 60; e += nt) {
    int i = e / 60, o = e % 60; float s = 0.f;
    for (int k = 0; k < 55; k++) s += sXa[i * 64 + k] * w2[k * 60 + o];
    sT[i * 64 + o] = s;
  }
  __syncthreads();
  for (int e = tid; e < 7 * 60; e += nt) {
    int i = e / 60, o = e % 60; float s = 0.f;
    for (int j = 0; j < 7; j++) s += sA[i * 7 + j] * sT[j * 64 + o];
    float v = seluf(s);
    sXa[i * 64 + o] = v; sXo[i * 64 + o] = v;
  }
  __syncthreads();
  cbam_lds(sXa, 64, 7, 60, m2w1, m2w2, m2c, 3, scr);
  for (int e = tid; e < 7 * 60; e += nt) {
    int i = e / 60, o = e % 60;
    out[b * 420 + e] = sXa[i * 64 + o] + sXo[i * 64 + o];
  }
}

// ---------------------------------------------------------------- host
struct GaP { const float *p,*bias,*q,*theta,*sw1,*sb1,*sw2,*sb2,*encw,*encb,*emb,*decw,*decb; };
struct RgP { const float *w1,*w2,*c1m1,*c1m2,*c1conv,*c2m1,*c2m2,*c2conv; };

static GaP readGa(void* const* d, int& i) {
  GaP g;
  g.p = (const float*)d[i++];    g.bias = (const float*)d[i++];
  g.q = (const float*)d[i++];    g.theta = (const float*)d[i++];
  g.sw1 = (const float*)d[i++];  g.sb1 = (const float*)d[i++];
  g.sw2 = (const float*)d[i++];  g.sb2 = (const float*)d[i++];
  g.encw = (const float*)d[i++]; g.encb = (const float*)d[i++];
  g.emb = (const float*)d[i++];  g.decw = (const float*)d[i++];
  g.decb = (const float*)d[i++];
  return g;
}
static RgP readRg(void* const* d, int& i) {
  RgP r;
  r.w1 = (const float*)d[i++];   r.w2 = (const float*)d[i++];
  r.c1m1 = (const float*)d[i++]; r.c1m2 = (const float*)d[i++]; r.c1conv = (const float*)d[i++];
  r.c2m1 = (const float*)d[i++]; r.c2m2 = (const float*)d[i++]; r.c2conv = (const float*)d[i++];
  return r;
}

extern "C" void kernel_launch(void* const* d_in, const int* in_sizes, int n_in,
                              void* d_out, int out_size, void* d_ws, size_t ws_size,
                              hipStream_t stream) {
  (void)in_sizes; (void)n_in; (void)out_size; (void)ws_size;
  const float* x = (const float*)d_in[0];
  int p = 1;
  GaP ga[8]; RgP rg[8];
  for (int r = 0; r < 7; r++) { ga[r] = readGa(d_in, p); rg[r] = readRg(d_in, p); }
  ga[7] = readGa(d_in, p); rg[7] = readRg(d_in, p);
  const float* wco = (const float*)d_in[p++];

  float* out = (float*)d_out;

  // workspace carve (256B aligned regions)
  char* wsb = (char*)d_ws; size_t off_b = 0;
  auto carve = [&](size_t bytes) -> void* {
    void* r = (void*)(wsb + off_b);
    off_b += (bytes + 255) & ~(size_t)255;
    return r;
  };
  float* gbuf   = (float*)carve(25600000UL * 4); // g / recon (aliased)
  bh*    abf    = (bh*)   carve(26214400UL * 2); // adj bf16 (GEMM A, Kpad rows)
  bh*    wbf    = (bh*)   carve(200704UL   * 2); // weight bf16 (N x Kpad, reused)
  float* ze     = (float*)carve(13107200UL * 4);
  bh*    zebf   = (bh*)   carve(13107200UL * 2); // Dpad stride
  float* scores = (float*)carve(52428800UL * 4);
  int*   idxb   = (int*)  carve(409600UL   * 4);
  float* e2     = (float*)carve(256UL      * 4);
  bh*    zqbf   = (bh*)   carve(13107200UL * 2); // Dpad stride
  float* adjout = (float*)carve(5120000UL  * 4);
  float* feats  = (float*)carve(31948800UL * 4);
  float* coarse = (float*)carve(2867200UL  * 4);

  const int B = 8192;
  k_zero1<<<1, 1, 0, stream>>>(out);

  const int SL_OFF[7] = {0, 5, 14, 23, 48, 57, 66};
  const int SL_C[7]   = {5, 9, 9, 25, 9, 9, 12};
  const int SL_D[7]   = {16, 32, 32, 256, 32, 32, 64};

  for (int r = 0; r < 7; r++) {
    const int c = SL_C[r], noff = SL_OFF[r], D = SL_D[r], Kcb = 32, f = 5, cc = c * c;
    const int ccPad = (cc + 31) & ~31;
    const int Dpad  = (D + 31) & ~31;
    const long M = (long)B * f; // multiple of 64
    k_ga_front<<<B, 256, 0, stream>>>(x, 390, noff, c, f, ga[r].p, ga[r].bias, ga[r].q, ga[r].theta, gbuf);
    k_transpose_norm<<<(unsigned)(B * f), 256, 0, stream>>>(gbuf, abf, c, f, ccPad);
    // encode: ze[M,D] = adj[M,ccPad] @ encw^T[D,ccPad] + encb
    k_cvt_w_t<<<128, 256, 0, stream>>>(ga[r].encw, wbf, cc, D, ccPad);
    dim3 g1((unsigned)(M / 64), (unsigned)(Dpad / 16));
    k_gemm_wmma<<<g1, 128, 0, stream>>>(abf, ccPad, wbf, ccPad, ga[r].encb,
                                        ze, D, zebf, Dpad, D, ccPad);
    // scores[M,Kcb] = ze[M,Dpad] @ emb[Kcb,Dpad]
    k_cvt_w_pad<<<32, 256, 0, stream>>>(ga[r].emb, wbf, D, Kcb, Dpad);
    dim3 g2((unsigned)(M / 64), (unsigned)((Kcb + 15) / 16));
    k_gemm_wmma<<<g2, 128, 0, stream>>>(zebf, Dpad, wbf, Dpad, nullptr,
                                        scores, Kcb, (bh*)nullptr, 0, Kcb, Dpad);
    k_e2<<<1, 256, 0, stream>>>(ga[r].emb, e2, Kcb, D);
    k_argmin<<<(unsigned)((M + 255) / 256), 256, 0, stream>>>(scores, e2, idxb, M, Kcb);
    float lscale = 1.25f / (float)(M * (long)D); // (1+BETA)*mean, region weight 1
    k_gather_loss<<<1024, 256, 0, stream>>>(ze, ga[r].emb, idxb, zqbf, out, lscale, M, D, Dpad);
    // recon[M,cc] = zq[M,Dpad] @ decw^T[cc,Dpad] + decb
    k_cvt_w_t<<<128, 256, 0, stream>>>(ga[r].decw, wbf, D, cc, Dpad);
    dim3 g3((unsigned)(M / 64), (unsigned)((cc + 15) / 16));
    k_gemm_wmma<<<g3, 128, 0, stream>>>(zqbf, Dpad, wbf, Dpad, ga[r].decb,
                                        gbuf, cc, (bh*)nullptr, 0, cc, Dpad);
    k_se_resid<<<B, 256, 0, stream>>>(gbuf, ga[r].sw1, ga[r].sb1, ga[r].sw2, ga[r].sb2, f, c, 2);
    k_adjout<<<B, 256, 0, stream>>>(gbuf, adjout, f, c);
    k_res_gcn<<<B, 128, 0, stream>>>(x, noff, c, adjout, rg[r].w1, rg[r].w2,
                                     rg[r].c1m1, rg[r].c1m2, rg[r].c1conv,
                                     rg[r].c2m1, rg[r].c2m2, rg[r].c2conv,
                                     (c / 2 > 0 ? c / 2 : 1), feats);
  }

  k_att_coarsen<<<B, 128, 0, stream>>>(feats, wco, coarse);

  { // coarse stage: f=50, c=7, K=128, D=32
    const int c = 7, D = 32, Kcb = 128, f = 50, cc = 49;
    const int ccPad = (cc + 31) & ~31; // 64
    const int Dpad  = (D + 31) & ~31;  // 32
    const long M = (long)B * f; // 409600, multiple of 64
    k_ga_front<<<B, 256, 0, stream>>>(coarse, 350, 0, c, f, ga[7].p, ga[7].bias, ga[7].q, ga[7].theta, gbuf);
    k_transpose_norm<<<(unsigned)((long)B * f), 256, 0, stream>>>(gbuf, abf, c, f, ccPad);
    k_cvt_w_t<<<8, 256, 0, stream>>>(ga[7].encw, wbf, cc, D, ccPad);
    dim3 g1((unsigned)(M / 64), (unsigned)(Dpad / 16));
    k_gemm_wmma<<<g1, 128, 0, stream>>>(abf, ccPad, wbf, ccPad, ga[7].encb,
                                        ze, D, zebf, Dpad, D, ccPad);
    k_cvt_w_pad<<<16, 256, 0, stream>>>(ga[7].emb, wbf, D, Kcb, Dpad);
    dim3 g2((unsigned)(M / 64), (unsigned)((Kcb + 15) / 16));
    k_gemm_wmma<<<g2, 128, 0, stream>>>(zebf, Dpad, wbf, Dpad, nullptr,
                                        scores, Kcb, (bh*)nullptr, 0, Kcb, Dpad);
    k_e2<<<1, 256, 0, stream>>>(ga[7].emb, e2, Kcb, D);
    k_argmin<<<(unsigned)((M + 255) / 256), 256, 0, stream>>>(scores, e2, idxb, M, Kcb);
    float lscale = 2.0f * 1.25f / (float)(M * (long)D); // co loss weighted 2.0
    k_gather_loss<<<1024, 256, 0, stream>>>(ze, ga[7].emb, idxb, zqbf, out, lscale, M, D, Dpad);
    k_cvt_w_t<<<8, 256, 0, stream>>>(ga[7].decw, wbf, D, cc, Dpad);
    dim3 g3((unsigned)(M / 64), (unsigned)((cc + 15) / 16));
    k_gemm_wmma<<<g3, 128, 0, stream>>>(zqbf, Dpad, wbf, Dpad, ga[7].decb,
                                        gbuf, cc, (bh*)nullptr, 0, cc, Dpad);
    k_se_resid<<<B, 256, 0, stream>>>(gbuf, ga[7].sw1, ga[7].sb1, ga[7].sw2, ga[7].sb2, f, c, 25);
    k_adjout<<<B, 256, 0, stream>>>(gbuf, adjout, f, c);
    k_res_gcn2<<<B, 128, 0, stream>>>(coarse, adjout, rg[7].w1, rg[7].w2,
                                      rg[7].c1m1, rg[7].c1m2, rg[7].c1conv,
                                      rg[7].c2m1, rg[7].c2m2, rg[7].c2conv, out + 1);
  }
}
